// Wav2Vec2GumbelVectorQuantizer_16578573763565
// MI455X (gfx1250) — matrix operand order repaired and checked by reading, not compiled, not run on hardware
//
#include <hip/hip_runtime.h>
#include <hip/hip_bf16.h>

// ---------------------------------------------------------------------------
// Wav2Vec2 Gumbel Vector Quantizer, fused for MI455X (gfx1250, wave32, WMMA)
//   B=8 S=2048 H=512  G=2 V=320 D=256  -> N=16384 tokens, GV=640 logits/token
// ---------------------------------------------------------------------------

typedef __attribute__((ext_vector_type(16))) __bf16 v16bf;
typedef __attribute__((ext_vector_type(8)))  float  v8f;

#define HDIM      512
#define GV        640
#define VCODES    320
#define NGROUPS   2
#define DHALF     128          // D / G
#define ROWS      16           // token rows per workgroup (one WMMA M-tile)
#define KC_COUNT  16           // 512 / 32 K-chunks
#define CT_COUNT  40           // 640 / 16 column tiles
#define CT_PER_WAVE 5          // 40 tiles / 8 waves

// round-to-nearest-even float -> bf16 (matches hardware cvt for normal values)
static __device__ __forceinline__ unsigned short f2bf(float x) {
    unsigned int u = __float_as_uint(x);
    unsigned int r = (u + 0x7FFFu + ((u >> 16) & 1u)) >> 16;
    return (unsigned short)r;
}

// ---------------------------------------------------------------------------
// Kernel 0: swizzle W_proj [512,640] fp32 -> bf16 B-fragments.
// Fragment (ct, kc) = 32x16 bf16 tile; per-lane 16 bf16 stored contiguously:
//   lane = (khalf<<4) | n ;  element 2v,2v+1 <-> K = 2v + 16*khalf (+1)
// (contiguous K order, same as the A fragments: lanes 0-15 hold K=0..15, 16-31 K=16..31)
// ---------------------------------------------------------------------------
__global__ void prep_w_kernel(const float* __restrict__ W,
                              unsigned short* __restrict__ wsW) {
    int idx = blockIdx.x * blockDim.x + threadIdx.x;   // lane-fragment id
    if (idx >= CT_COUNT * KC_COUNT * 32) return;
    int lane  = idx & 31;
    int frag  = idx >> 5;              // ct*16 + kc
    int kc    = frag & 15;
    int ct    = frag >> 4;
    int col   = ct * 16 + (lane & 15);
    int khalf = lane >> 4;
    unsigned short* dst = wsW + (size_t)frag * 512 + lane * 16;
#pragma unroll
    for (int v = 0; v < 8; ++v) {
        int k = kc * 32 + khalf * 16 + 2 * v;
        dst[2 * v]     = f2bf(W[(size_t)k * GV + col]);
        dst[2 * v + 1] = f2bf(W[(size_t)(k + 1) * GV + col]);
    }
}

// ---------------------------------------------------------------------------
// Kernel 1: fused GEMM (bf16 WMMA) + bias + gumbel argmax + codevector gather
//           + deterministic per-block marginal partials.
// ---------------------------------------------------------------------------
__global__ __launch_bounds__(256) void vq_main_kernel(
    const float* __restrict__ hidden,          // [N,512]
    const unsigned char* __restrict__ mask,    // [N] (jnp bool, 1 byte)
    const float* __restrict__ bias,            // [640]
    const float* __restrict__ codevectors,     // [640,128]
    const float* __restrict__ gumbels,         // [N*2,320]
    const unsigned short* __restrict__ wsW,    // swizzled bf16 W fragments
    float* __restrict__ out,                   // [N,256] codevecs
    float* __restrict__ partials,              // [nblocks,640]
    float* __restrict__ masksums)              // [nblocks]
{
    __shared__ __align__(32) unsigned short fragA[KC_COUNT * 32 * 16]; // 16 KB
    __shared__ float slog[ROWS * GV];                                  // 40 KB
    __shared__ float lmaxS[32];
    __shared__ float linvS[32];
    __shared__ float mrow[ROWS];

    const int tid  = threadIdx.x;
    const int bid  = blockIdx.x;
    const int r0   = bid * ROWS;
    const int lane = tid & 31;
    const int wave = tid >> 5;

    // Prefetch this block's gumbel window (16 rows x 2 groups x 320 f32 =
    // 40960 B) so the phase-4 stream hits cache after the GEMM barrier.
    // global_prefetch_b8 uses no counters / registers; overlaps whole GEMM.
    {
        const char* gbase = (const char*)(gumbels +
                            (size_t)r0 * NGROUPS * VCODES);
        __builtin_prefetch(gbase + tid * 128, 0, 1);
        if (tid < 64) __builtin_prefetch(gbase + 32768 + tid * 128, 0, 1);
    }

    if (tid < ROWS) mrow[tid] = mask[r0 + tid] ? 1.0f : 0.0f;

    // ---- Phase 1: build A fragments (16x32 bf16 per kc) in LDS ----
    // layout: lanes 0-15 M=0..15 {V0:K0,1 .. V7:K14,15},
    //         lanes 16-31 same M  {V0:K16,17 .. V7:K30,31}
#pragma unroll
    for (int rep = 0; rep < 2; ++rep) {
        int f    = tid + rep * 256;          // 0..511 lane-fragment id
        int l    = f & 31;
        int kc   = f >> 5;
        int M    = l & 15, half = l >> 4;
        const float* src = hidden + (size_t)(r0 + M) * HDIM + kc * 32;
        unsigned short* dst = fragA + kc * 512 + l * 16;
#pragma unroll
        for (int v = 0; v < 8; ++v) {
            int kb = 2 * v + 16 * half;
            float2 hv = *(const float2*)(src + kb);
            dst[2 * v]     = f2bf(hv.x);
            dst[2 * v + 1] = f2bf(hv.y);
        }
    }
    __syncthreads();

    // ---- Phase 2: GEMM. Each wave owns 5 column tiles of 16 cols ----
    v8f zero = {0.f, 0.f, 0.f, 0.f, 0.f, 0.f, 0.f, 0.f};
    v8f acc[CT_PER_WAVE];
#pragma unroll
    for (int c = 0; c < CT_PER_WAVE; ++c) acc[c] = zero;

    for (int kc = 0; kc < KC_COUNT; ++kc) {
        v16bf a = *(const v16bf*)(fragA + kc * 512 + lane * 16);
#pragma unroll
        for (int c = 0; c < CT_PER_WAVE; ++c) {
            int ctg = wave * CT_PER_WAVE + c;
            v16bf b = *(const v16bf*)(wsW + ((size_t)ctg * KC_COUNT + kc) * 512 + lane * 16);
            acc[c] = __builtin_amdgcn_wmma_f32_16x16x32_bf16(
                false, a, false, b, (short)0, acc[c], false, false);
        }
    }

    // ---- Phase 3: logits (+bias) -> LDS. C/D layout: row = r + 8*(lane>=16)
#pragma unroll
    for (int c = 0; c < CT_PER_WAVE; ++c) {
        int ctg   = wave * CT_PER_WAVE + c;
        int col   = ctg * 16 + (lane & 15);
        float bcol = bias[col];
        int rbase = (lane >> 4) * 8;
#pragma unroll
        for (int r = 0; r < 8; ++r)
            slog[(rbase + r) * GV + col] = acc[c][r] + bcol;
    }
    __syncthreads();

    // ---- Phase 4: per (row,group): gumbel argmax, softmax stats, gather ----
#pragma unroll
    for (int i = 0; i < 4; ++i) {
        int p   = wave + 8 * i;        // 0..31 unique (row,group) pairs
        int row = p >> 1;
        int g   = p & 1;
        const float* gptr = gumbels + ((size_t)(r0 + row) * NGROUPS + g) * VCODES;
        float vals[10];
        float zb = -1e30f; int ib = 0;
        float lm = -1e30f;
#pragma unroll
        for (int j = 0; j < 10; ++j) {
            int cv   = lane + 32 * j;
            float lg = slog[row * GV + g * VCODES + cv];
            vals[j]  = lg;
            lm = fmaxf(lm, lg);
            float z = lg + gptr[cv];          // argmax of softmax(z/tau) == argmax z
            if (z > zb) { zb = z; ib = cv; }
        }
        for (int off = 16; off > 0; off >>= 1) {
            float oz = __shfl_xor(zb, off);
            int   oi = __shfl_xor(ib, off);
            if (oz > zb || (oz == zb && oi < ib)) { zb = oz; ib = oi; }
            lm = fmaxf(lm, __shfl_xor(lm, off));
        }
        float se = 0.f;
#pragma unroll
        for (int j = 0; j < 10; ++j) se += __expf(vals[j] - lm);
        for (int off = 16; off > 0; off >>= 1) se += __shfl_xor(se, off);
        if (lane == 0) { lmaxS[p] = lm; linvS[p] = 1.0f / se; }

        // one-hot codebook combine == gather codevector at argmax
        const float4* cb = (const float4*)(codevectors + ((size_t)g * VCODES + ib) * DHALF);
        float4* op = (float4*)(out + (size_t)(r0 + row) * (NGROUPS * DHALF) + g * DHALF);
        op[lane] = cb[lane];
    }
    __syncthreads();

    // ---- Phase 5: deterministic per-block masked marginal partials ----
    for (int c = tid; c < GV; c += 256) {
        int g = (c >= VCODES) ? 1 : 0;
        float s = 0.f;
#pragma unroll
        for (int row = 0; row < ROWS; ++row) {
            int p = row * 2 + g;
            s += mrow[row] * __expf(slog[row * GV + c] - lmaxS[p]) * linvS[p];
        }
        partials[(size_t)bid * GV + c] = s;
    }
    if (tid == 0) {
        float ms = 0.f;
        for (int r = 0; r < ROWS; ++r) ms += mrow[r];
        masksums[bid] = ms;
    }
}

// ---------------------------------------------------------------------------
// Kernel 2: fixed-order reduction of partials -> marginal -> perplexity
// ---------------------------------------------------------------------------
__global__ void finalize_kernel(const float* __restrict__ partials,
                                const float* __restrict__ masksums,
                                float* __restrict__ out, int nwg,
                                unsigned long long ppl_index) {
    __shared__ float marg[GV];
    __shared__ float msS;
    __shared__ float e0s[256], e1s[256];
    int t = threadIdx.x;
    for (int c = t; c < GV; c += 256) {
        float s = 0.f;
        for (int w = 0; w < nwg; ++w) s += partials[(size_t)w * GV + c];
        marg[c] = s;
    }
    if (t == 0) {
        float ms = 0.f;
        for (int w = 0; w < nwg; ++w) ms += masksums[w];
        msS = ms;
    }
    __syncthreads();
    float inv = 1.0f / msS;
    float e0 = 0.f, e1 = 0.f;
    for (int c = t; c < GV; c += 256) {
        float mg = marg[c] * inv;
        float term = -mg * logf(mg + 1e-7f);
        if (c < VCODES) e0 += term; else e1 += term;
    }
    e0s[t] = e0; e1s[t] = e1;
    __syncthreads();
    if (t == 0) {
        float h0 = 0.f, h1 = 0.f;
        for (int i = 0; i < 256; ++i) { h0 += e0s[i]; h1 += e1s[i]; }
        out[ppl_index] = __expf(h0) + __expf(h1);
    }
}

// ---------------------------------------------------------------------------
extern "C" void kernel_launch(void* const* d_in, const int* in_sizes, int n_in,
                              void* d_out, int out_size, void* d_ws, size_t ws_size,
                              hipStream_t stream) {
    const float*         hidden      = (const float*)d_in[0];
    const unsigned char* mask        = (const unsigned char*)d_in[1];
    const float*         W           = (const float*)d_in[2];
    const float*         bias        = (const float*)d_in[3];
    const float*         codevectors = (const float*)d_in[4];
    const float*         gumbels     = (const float*)d_in[5];
    float*               out         = (float*)d_out;

    const int N  = in_sizes[1];     // B*S = 16384
    const int NB = N / ROWS;        // 1024 workgroups

    // workspace layout: [bf16 swizzled W 640KB][partials NB*640 f32][masksums NB f32]
    unsigned short* wsW = (unsigned short*)d_ws;
    float* partials = (float*)((char*)d_ws + (size_t)CT_COUNT * KC_COUNT * 512 * 2);
    float* masksums = partials + (size_t)NB * GV;

    prep_w_kernel<<<(CT_COUNT * KC_COUNT * 32 + 255) / 256, 256, 0, stream>>>(W, wsW);
    vq_main_kernel<<<NB, 256, 0, stream>>>(hidden, mask, bias, codevectors, gumbels,
                                           wsW, out, partials, masksums);
    finalize_kernel<<<1, 256, 0, stream>>>(partials, masksums, out, NB,
                                           (unsigned long long)N * (NGROUPS * DHALF));
}
